// EdgeSAGELayer_59021440582266
// MI455X (gfx1250) — matrix operand order repaired and checked by reading, not compiled, not run on hardware
//
#include <hip/hip_runtime.h>
#include <hip/hip_bf16.h>

typedef _Float16 v16h __attribute__((ext_vector_type(16)));
typedef _Float16 h8   __attribute__((ext_vector_type(8)));
typedef float    v8f  __attribute__((ext_vector_type(8)));

#define D 64  // D_IN == D_OUT == 64

// ---------------------------------------------------------------------------
// Kernel 1: zero the scatter accumulators (agg[N*64] and cnt[N])
// ---------------------------------------------------------------------------
__global__ void zero_kernel(float* p, int n) {
    int i = blockIdx.x * blockDim.x + threadIdx.x;
    if (i < n) p[i] = 0.0f;
}

// ---------------------------------------------------------------------------
// Kernel 2: scatter-add edge_attr onto src nodes + per-node counts.
// One thread per (edge, feature). f32 atomics resolve in L2.
// ---------------------------------------------------------------------------
__global__ void scatter_kernel(const float* __restrict__ edge_attr,
                               const int* __restrict__ edge_index,
                               float* __restrict__ agg,
                               float* __restrict__ cnt,
                               int E) {
    long long tid = (long long)blockIdx.x * blockDim.x + threadIdx.x;
    if (tid >= (long long)E * D) return;
    int e = (int)(tid >> 6);
    int d = (int)(tid & (D - 1));
    int s = edge_index[e];  // src row
    unsafeAtomicAdd(&agg[(long long)s * D + d], edge_attr[tid]);
    if (d == 0) unsafeAtomicAdd(&cnt[s], 1.0f);
}

// ---------------------------------------------------------------------------
// Kernel 3: node embeddings via WMMA.
//   x = node_attr + 0.5 * agg / max(cnt,1)   (built in LDS as f16)
//   emb = sigmoid(x @ W^T + b)
// Block = 128 threads (4 waves). Block covers 16 nodes; wave w covers output
// cols [16w, 16w+16). K-loop: two v_wmma_f32_16x16x32_f16 steps.
// N is a multiple of 16 (50000 = 3125*16) -> EXEC all-ones at every WMMA.
// Sigmoid epilogue uses v_exp_f32 + v_rcp_f32 (native transcendentals).
// ---------------------------------------------------------------------------
__global__ void node_kernel(const float* __restrict__ node_attr,
                            const float* __restrict__ agg,
                            const float* __restrict__ cnt,
                            const float* __restrict__ W,
                            const float* __restrict__ bias,
                            float* __restrict__ emb) {
    __shared__ _Float16 xs[16][D];   // x tile, f16        (2 KB)
    __shared__ _Float16 wh[D][D];    // W in f16, row-major (8 KB)

    const int t = threadIdx.x;
    const int node0 = blockIdx.x * 16;

    // Stage W (64x64 f32 -> f16): 4096 elems / 128 threads
    for (int i = t; i < D * D; i += 128)
        (&wh[0][0])[i] = (_Float16)W[i];

    // Stage x tile: 16*64 elems / 128 threads
    for (int i = t; i < 16 * D; i += 128) {
        int r = i >> 6, k = i & (D - 1);
        int node = node0 + r;
        float c = cnt[node];
        if (c < 1.0f) c = 1.0f;
        float x = node_attr[(long long)node * D + k]
                + 0.5f * agg[(long long)node * D + k] / c;
        xs[r][k] = (_Float16)x;
    }
    __syncthreads();

    const int wave = t >> 5;
    const int lane = t & 31;
    const int hi   = lane >> 4;    // lane group selector
    const int lr   = lane & 15;    // row (A) / col (B,C,D) within tile
    const int col  = wave * 16 + lr;

    // Accumulator seeded with bias: c[r] -> out[M = r + 8*hi][N = col]
    v8f acc;
    {
        float bv = bias[col];
#pragma unroll
        for (int r = 0; r < 8; ++r) acc[r] = bv;
    }

#pragma unroll
    for (int k0 = 0; k0 < D; k0 += 32) {
        // A fragment (16x32 f16): lanes<16 hold K={k0..k0+7, k0+16..k0+23},
        // lanes>=16 hold K shifted by 8. Row = lr.
        h8 alo = *(const h8*)&xs[lr][k0 + hi * 8];
        h8 ahi = *(const h8*)&xs[lr][k0 + 16 + hi * 8];
        v16h a;
#pragma unroll
        for (int j = 0; j < 8; ++j) { a[j] = alo[j]; a[j + 8] = ahi[j]; }

        // B fragment (32x16 f16): B[k][n] = W[n][k]. Lane holds col=col,
        // K = {k0+8*hi..+7, k0+16+8*hi..+7} (same order as A) in wh[col][..]
        int kb = k0 + hi * 8;
        h8 b0 = *(const h8*)&wh[col][kb];
        h8 b1 = *(const h8*)&wh[col][kb + 16];
        v16h b;
#pragma unroll
        for (int j = 0; j < 8; ++j) { b[j] = b0[j]; b[j + 8] = b1[j]; }

        acc = __builtin_amdgcn_wmma_f32_16x16x32_f16(
            /*neg_a=*/false, a, /*neg_b=*/false, b,
            /*c_mod=*/(short)0, acc, /*reuse_a=*/false, /*reuse_b=*/false);
    }

    // Write sigmoid(acc): c[r] -> node (node0 + r + 8*hi), column col.
    // sigmoid = rcp(1 + exp(-x)) via native v_exp_f32 / v_rcp_f32.
#pragma unroll
    for (int r = 0; r < 8; ++r) {
        int node = node0 + r + hi * 8;
        float v = acc[r];
        v = __builtin_amdgcn_rcpf(1.0f + __expf(-v));
        emb[(long long)node * D + col] = v;
    }
}

// ---------------------------------------------------------------------------
// Kernel 4: edge embeddings = 0.5*(emb[src] + emb[dst]).
// emb is 12.8 MB -> L2 resident; output stores coalesced float4.
// One thread per 4 output floats.
// ---------------------------------------------------------------------------
__global__ void edge_kernel(const float* __restrict__ emb,
                            const int* __restrict__ edge_index,
                            float* __restrict__ out,
                            int E) {
    long long tid = (long long)blockIdx.x * blockDim.x + threadIdx.x;
    if (tid >= (long long)E * (D / 4)) return;
    int e = (int)(tid >> 4);          // D/4 == 16 quads per edge
    int q = (int)(tid & 15) * 4;
    int s = edge_index[e];
    int d = edge_index[E + e];
    const float4 a = *(const float4*)&emb[(long long)s * D + q];
    const float4 b = *(const float4*)&emb[(long long)d * D + q];
    float4 o;
    o.x = 0.5f * (a.x + b.x);
    o.y = 0.5f * (a.y + b.y);
    o.z = 0.5f * (a.z + b.z);
    o.w = 0.5f * (a.w + b.w);
    *(float4*)&out[tid * 4] = o;
}

// ---------------------------------------------------------------------------
extern "C" void kernel_launch(void* const* d_in, const int* in_sizes, int n_in,
                              void* d_out, int out_size, void* d_ws, size_t ws_size,
                              hipStream_t stream) {
    const float* edge_attr  = (const float*)d_in[0];
    const int*   edge_index = (const int*)d_in[1];
    const float* node_attr  = (const float*)d_in[2];
    const float* W          = (const float*)d_in[3];
    const float* b          = (const float*)d_in[4];
    float*       out        = (float*)d_out;

    const int E = in_sizes[0] / D;   // 800000
    const int N = in_sizes[2] / D;   // 50000

    // Workspace layout (floats): agg[N*D] | cnt[N] | emb[N*D]
    float* agg = (float*)d_ws;
    float* cnt = agg + (size_t)N * D;
    float* emb = cnt + (size_t)N;

    // 1) zero agg + cnt (N*D + N contiguous floats)
    {
        int n = N * D + N;
        zero_kernel<<<(n + 255) / 256, 256, 0, stream>>>(agg, n);
    }
    // 2) scatter-add
    {
        long long n = (long long)E * D;
        int blocks = (int)((n + 255) / 256);
        scatter_kernel<<<blocks, 256, 0, stream>>>(edge_attr, edge_index, agg, cnt, E);
    }
    // 3) node embeddings (WMMA). N is a multiple of 16.
    node_kernel<<<N / 16, 128, 0, stream>>>(node_attr, agg, cnt, W, b, emb);
    // 4) edge embeddings
    {
        long long n = (long long)E * (D / 4);
        int blocks = (int)((n + 255) / 256);
        edge_kernel<<<blocks, 256, 0, stream>>>(emb, edge_index, out, E);
    }
}
